// FusionTransformer_77738908058294
// MI455X (gfx1250) — compile-verified, matrix operand order checked by reading, not run on hardware
//
#include <hip/hip_runtime.h>
#include <hip/hip_bf16.h>
#include <math.h>

// ---------------------------------------------------------------------------
// FusionTransformer cross-attention for MI455X (gfx1250, wave32, WMMA).
//   s,g : [B=4, V=25, C=108, H=64, W=64] f32
// Prep kernel: repack the 8 weight matrices (f32 108x108) into bf16 WMMA
//   A-fragment layout in d_ws (zero-padded to 112x128) -> hot kernel loads
//   each A fragment with ONE aligned 32-byte global load (L2/WGP$ resident).
// Pass kernel: 1 wave per (batch-group, 16-pixel tile). Activations staged
//   to LDS in B-fragment layout with coalesced unconditional loads; q/k/v
//   projections and the output projection run on v_wmma_f32_16x16x32_bf16.
// Register budget capped (launch_bounds) so occupancy is LDS-limited, not
//   VGPR-limited: memory-bound kernel -> want concurrent waves per SIMD.
// ---------------------------------------------------------------------------

typedef __bf16 bf16;
typedef __bf16 v16bf __attribute__((ext_vector_type(16)));
typedef float  v8f   __attribute__((ext_vector_type(8)));

#define C_DIM 108
#define HW    4096           // 64*64
#define NVIEW 5
#define TILE  16             // pixels per block (WMMA N)
#define MT    7              // ceil(108/16) output-channel tiles
#define KT    4              // ceil(108/32) K-steps (padded to 128)
#define FRAGS_PER_MAT (MT * KT)              // 28 fragments / matrix
#define ELEMS_PER_MAT (FRAGS_PER_MAT * 512)  // 14336 bf16 / matrix

// element e (0..15) of a 16-bit WMMA A/B fragment maps to K offset:
//   K = 16*(e>>3) + 8*hi + (e&7)    (hi = lane>>4)
__device__ __forceinline__ int frag_k(int e, int hi) {
    return ((e >> 3) << 4) + (hi << 3) + (e & 7);
}

__device__ __forceinline__ v8f wmma_bf16(v16bf a, v16bf b, v8f c) {
    return __builtin_amdgcn_wmma_f32_16x16x32_bf16(false, a, false, b,
                                                   (short)0, c, false, false);
}

// ---------------------------------------------------------------------------
// Prep: 8 matrices x 28 fragments; block = one fragment (32 lanes x 16 bf16).
// ---------------------------------------------------------------------------
__global__ __launch_bounds__(32)
void prep_weight_frags(const float* __restrict__ W0, const float* __restrict__ W1,
                       const float* __restrict__ W2, const float* __restrict__ W3,
                       const float* __restrict__ W4, const float* __restrict__ W5,
                       const float* __restrict__ W6, const float* __restrict__ W7,
                       bf16* __restrict__ out)
{
    const int mat = blockIdx.x / FRAGS_PER_MAT;
    const int fi  = blockIdx.x % FRAGS_PER_MAT;
    const int mt  = fi >> 2, kt = fi & 3;
    const int lane = threadIdx.x, p = lane & 15, hi = lane >> 4;

    const float* W = (mat == 0) ? W0 : (mat == 1) ? W1 : (mat == 2) ? W2 :
                     (mat == 3) ? W3 : (mat == 4) ? W4 : (mat == 5) ? W5 :
                     (mat == 6) ? W6 : W7;

    v16bf frag;
    #pragma unroll
    for (int e = 0; e < 16; ++e) {
        int d = mt * 16 + p;                 // A row = lane&15
        int k = kt * 32 + frag_k(e, hi);
        float w = (d < C_DIM && k < C_DIM) ? W[d * C_DIM + k] : 0.f;
        frag[e] = (bf16)w;
    }
    *(v16bf*)&out[((size_t)blockIdx.x * 32 + lane) * 16] = frag;
}

// ---------------------------------------------------------------------------
__global__ __launch_bounds__(32, 4)     // cap ~256 VGPRs: no vgpr-msb churn
void fusion_attn_pass(const float* __restrict__ S,   // query-side input (+residual)
                      const float* __restrict__ G,   // key/value-side input
                      float* __restrict__ OUT,
                      const bf16* __restrict__ WqF, const bf16* __restrict__ WkF,
                      const bf16* __restrict__ WvF, const bf16* __restrict__ WoF,
                      const float* __restrict__ bq, const float* __restrict__ bk,
                      const float* __restrict__ bv, const float* __restrict__ bo,
                      int va, int fb)                // linear view = v*va + f*fb
{
    // q/k/v results: [proj][view][chan(112)][pixel] bf16        = 53760 B
    __shared__ bf16 sQKV[3][NVIEW][112][TILE];
    // staging buffer in exact WMMA B-fragment layout (32 lanes x 16 bf16/kt)
    __shared__ __align__(32) bf16 sFrag[KT][512];               //  4096 B
    // biases, zero-padded to 112                                 //  1792 B
    __shared__ __align__(32) float sBias[4][112];
    // per-pixel attention weights a[n][m]                        //  1600 B
    __shared__ float sA[NVIEW][NVIEW][TILE];

    const int lane = threadIdx.x;     // 0..31
    const int p    = lane & 15;       // WMMA column / pixel-in-tile & A-row
    const int hi   = lane >> 4;

    const int gidx    = blockIdx.x >> 8;    // 0..19 : batch*5 + fixed(row|col)
    const int pixbase = (blockIdx.x & 255) * TILE;
    const int b = gidx / 5;
    const int f = gidx % 5;

    // warm WGP$/L2 with the shared fragment-layout weights
    __builtin_prefetch(WqF + lane * 448, 0, 0);
    __builtin_prefetch(WkF + lane * 448, 0, 0);
    __builtin_prefetch(WvF + lane * 448, 0, 0);
    __builtin_prefetch(WoF + lane * 448, 0, 0);

    // ---- one-time LDS init: biases + zero K-pad rows (108..127) of sFrag ---
    for (int i = lane; i < 4 * 112; i += 32) {
        int pr = i / 112, d = i % 112;
        const float* bp = (pr == 0) ? bq : (pr == 1) ? bk : (pr == 2) ? bv : bo;
        sBias[pr][d] = (d < C_DIM) ? bp[d] : 0.f;
    }
    for (int i = lane; i < (128 - C_DIM) * TILE; i += 32) {
        int K = C_DIM + (i >> 4), col = i & 15;
        int kt = K >> 5, Kin = K & 31;
        int tl = col + (((Kin >> 3) & 1) << 4);
        int e  = ((Kin >> 4) << 3) | (Kin & 7);
        sFrag[kt][tl * 16 + e] = (bf16)0.f;   // persists: staging writes c<108 only
    }
    __syncthreads();

    // stage one view's [108 x 16] f32 tile into sFrag as bf16 B-fragments;
    // fully coalesced: wave reads 2 channel rows (128 B) per iteration.
    auto stage_tile = [&](const float* __restrict__ X, long vb) {
        for (int i = 0; i < 54; ++i) {
            int c = 2 * i + hi;
            float x = X[vb + (long)c * HW + pixbase + p];
            int kt = c >> 5, Kin = c & 31;
            int tl = p + (((Kin >> 3) & 1) << 4);
            int e  = ((Kin >> 4) << 3) | (Kin & 7);
            sFrag[kt][tl * 16 + e] = (bf16)x;
        }
    };

    // D = W x B (+bias) for one (proj, view); A frags = 32B global loads.
    auto gemm_store = [&](const bf16* __restrict__ WF, int pr, int v,
                          const v16bf* bfr) {
        for (int mt = 0; mt < MT; ++mt) {
            v8f acc = *(const v8f*)&sBias[pr][mt * 16 + hi * 8];
            #pragma unroll
            for (int kt = 0; kt < KT; ++kt) {
                v16bf afrag = *(const v16bf*)&WF[((mt * 4 + kt) * 32 + lane) * 16];
                acc = wmma_bf16(afrag, bfr[kt], acc);
            }
            #pragma unroll
            for (int r = 0; r < 8; ++r)
                sQKV[pr][v][mt * 16 + r + hi * 8][p] = (bf16)acc[r];
        }
    };

    // ---------------- Phase 1: q/k/v projections (WMMA) --------------------
    #pragma unroll 1
    for (int v = 0; v < NVIEW; ++v) {
        const long vb = (long)(b * 25 + v * va + f * fb) * (C_DIM * HW);

        stage_tile(S, vb);
        __syncthreads();
        v16bf bS[KT];
        #pragma unroll
        for (int kt = 0; kt < KT; ++kt)
            bS[kt] = *(const v16bf*)&sFrag[kt][lane * 16];
        __syncthreads();                       // sFrag free for G

        stage_tile(G, vb);
        __syncthreads();
        v16bf bG[KT];
        #pragma unroll
        for (int kt = 0; kt < KT; ++kt)
            bG[kt] = *(const v16bf*)&sFrag[kt][lane * 16];

        gemm_store(WqF, 0, v, bS);             // q = Wq s + bq
        gemm_store(WkF, 1, v, bG);             // k = Wk g + bk
        gemm_store(WvF, 2, v, bG);             // v = Wv g + bv
        __syncthreads();                       // before next view reuses sFrag
    }

    // ---------------- Phase 2: per-pixel 5x5 scores + softmax --------------
    // lanes split the 108 channels (hi=0: 0..53, hi=1: 54..107); partial dots
    // combined across halves with a lane-16 xor shuffle (wave32).
    {
        float dot[NVIEW][NVIEW];
        #pragma unroll
        for (int n = 0; n < NVIEW; ++n)
            #pragma unroll
            for (int m = 0; m < NVIEW; ++m) dot[n][m] = 0.f;

        for (int c = 0; c < 54; ++c) {
            int cc = c + hi * 54;
            float qv[NVIEW], kv[NVIEW];
            #pragma unroll
            for (int n = 0; n < NVIEW; ++n) qv[n] = (float)sQKV[0][n][cc][p];
            #pragma unroll
            for (int m = 0; m < NVIEW; ++m) kv[m] = (float)sQKV[1][m][cc][p];
            #pragma unroll
            for (int n = 0; n < NVIEW; ++n)
                #pragma unroll
                for (int m = 0; m < NVIEW; ++m)
                    dot[n][m] = fmaf(qv[n], kv[m], dot[n][m]);
        }
        const float scale = rsqrtf((float)C_DIM);
        #pragma unroll
        for (int n = 0; n < NVIEW; ++n) {
            float row[NVIEW];
            float mx = -3.0e38f;
            #pragma unroll
            for (int m = 0; m < NVIEW; ++m) {
                float full = dot[n][m] + __shfl_xor(dot[n][m], 16, 32);
                row[m] = full * scale;
                mx = fmaxf(mx, row[m]);
            }
            float sum = 0.f;
            #pragma unroll
            for (int m = 0; m < NVIEW; ++m) { row[m] = __expf(row[m] - mx); sum += row[m]; }
            float inv = 1.f / sum;
            if (hi == 0) {                     // both halves hold identical values
                #pragma unroll
                for (int m = 0; m < NVIEW; ++m) sA[n][m][p] = row[m] * inv;
            }
        }
    }

    // ---------------- Phase 3: per view, o = attn*v, out = Wo*o + bo + s ---
    #pragma unroll 1
    for (int n = 0; n < NVIEW; ++n) {
        __syncthreads();                       // sA ready / sFrag free
        float an[NVIEW];
        #pragma unroll
        for (int m = 0; m < NVIEW; ++m) an[m] = sA[n][m][p];

        for (int c = 0; c < 54; ++c) {         // write o in B-fragment layout
            int cc = c + hi * 54;
            float o = 0.f;
            #pragma unroll
            for (int m = 0; m < NVIEW; ++m)
                o = fmaf(an[m], (float)sQKV[2][m][cc][p], o);
            int kt = cc >> 5, Kin = cc & 31;
            int tl = p + (((Kin >> 3) & 1) << 4);
            int e  = ((Kin >> 4) << 3) | (Kin & 7);
            sFrag[kt][tl * 16 + e] = (bf16)o;
        }
        __syncthreads();

        const long vb = (long)(b * 25 + n * va + f * fb) * (C_DIM * HW);
        for (int mt = 0; mt < MT; ++mt) {
            v8f acc = *(const v8f*)&sBias[3][mt * 16 + hi * 8];
            #pragma unroll
            for (int kt = 0; kt < KT; ++kt) {
                v16bf afrag = *(const v16bf*)&WoF[((mt * 4 + kt) * 32 + lane) * 16];
                v16bf bfrag = *(const v16bf*)&sFrag[kt][lane * 16];
                acc = wmma_bf16(afrag, bfrag, acc);
            }
            #pragma unroll
            for (int r = 0; r < 8; ++r) {
                int d = mt * 16 + r + hi * 8;
                if (d < C_DIM) {
                    long idx = vb + (long)d * HW + pixbase + p;
                    OUT[idx] = acc[r] + S[idx];    // residual
                }
            }
        }
    }
}

// ---------------------------------------------------------------------------
extern "C" void kernel_launch(void* const* d_in, const int* in_sizes, int n_in,
                              void* d_out, int out_size, void* d_ws, size_t ws_size,
                              hipStream_t stream) {
    (void)in_sizes; (void)n_in; (void)out_size; (void)ws_size;

    const float* s   = (const float*)d_in[0];
    const float* g   = (const float*)d_in[1];
    const float* WqR = (const float*)d_in[2];
    const float* WkR = (const float*)d_in[3];
    const float* WvR = (const float*)d_in[4];
    const float* WoR = (const float*)d_in[5];
    const float* WqC = (const float*)d_in[6];
    const float* WkC = (const float*)d_in[7];
    const float* WvC = (const float*)d_in[8];
    const float* WoC = (const float*)d_in[9];
    const float* bqR = (const float*)d_in[10];
    const float* bkR = (const float*)d_in[11];
    const float* bvR = (const float*)d_in[12];
    const float* boR = (const float*)d_in[13];
    const float* bqC = (const float*)d_in[14];
    const float* bkC = (const float*)d_in[15];
    const float* bvC = (const float*)d_in[16];
    const float* boC = (const float*)d_in[17];

    float* out = (float*)d_out;
    bf16* wf   = (bf16*)d_ws;          // 8 * 14336 bf16 = 224 KiB

    // Repack all 8 weight matrices into WMMA A-fragment layout (bf16).
    prep_weight_frags<<<dim3(8 * FRAGS_PER_MAT), dim3(32), 0, stream>>>(
        WqR, WkR, WvR, WoR, WqC, WkC, WvC, WoC, wf);

    const bf16* WqFr = wf + 0 * ELEMS_PER_MAT;
    const bf16* WkFr = wf + 1 * ELEMS_PER_MAT;
    const bf16* WvFr = wf + 2 * ELEMS_PER_MAT;
    const bf16* WoFr = wf + 3 * ELEMS_PER_MAT;
    const bf16* WqFc = wf + 4 * ELEMS_PER_MAT;
    const bf16* WkFc = wf + 5 * ELEMS_PER_MAT;
    const bf16* WvFc = wf + 6 * ELEMS_PER_MAT;
    const bf16* WoFc = wf + 7 * ELEMS_PER_MAT;

    // 20 batch-groups (B*5) x 256 pixel tiles; 1 wave per block
    dim3 grid(20 * 256), block(32);

    // Row pass: group = (b, row i), attended index = col j, linear = j + i*5.
    fusion_attn_pass<<<grid, block, 0, stream>>>(
        s, g, out, WqFr, WkFr, WvFr, WoFr, bqR, bkR, bvR, boR, /*va=*/1, /*fb=*/5);

    // Col pass: group = (b, col j), attended index = row i, linear = i*5 + j.
    // In-place on d_out: each block reads only the 5 views it rewrites.
    fusion_attn_pass<<<grid, block, 0, stream>>>(
        out, g, out, WqFc, WkFc, WvFc, WoFc, bqC, bkC, bvC, boC, /*va=*/5, /*fb=*/1);
}
